// KeypointSampler_3831110828402
// MI455X (gfx1250) — compile-verified, matrix operand order checked by reading, not run on hardware
//
#include <hip/hip_runtime.h>
#include <float.h>

typedef __attribute__((ext_vector_type(16))) _Float16 v16h;
typedef __attribute__((ext_vector_type(8)))  float    v8f;
typedef __attribute__((ext_vector_type(4)))  float    f4;
typedef __attribute__((ext_vector_type(2)))  float    f2;

#define IMG    2048
#define NTILES (8 * 256 * 256)        // 524288 tiles of 8x8

#define LOG2E 1.44269504088896340736f
#define LN2   0.69314718055994530942f

__device__ __forceinline__ float mix_u01(unsigned h) {
    h *= 0x9E3779B1u;
    h ^= h >> 15; h *= 0x85EBCA77u; h ^= h >> 13;
    return __builtin_fmaf((float)h, 0x1p-32f, 0x1p-33f);   // (0,1)
}

__global__ __launch_bounds__(256) void keypoint_sampler_kernel(
        const float* __restrict__ x, float* __restrict__ out) {
    const int tid  = threadIdx.x;
    const int lane = tid & 31;
    const int wave = tid >> 5;
    const int m    = lane & 15;       // tile == WMMA A-matrix row
    const int half = lane >> 4;       // which 4 columns of the tile this lane owns
    const int h4   = half * 4;

    const int t  = blockIdx.x * 128 + wave * 16 + m;
    const int wc = t & 255;
    const int hc = (t >> 8) & 255;
    const int b  = t >> 16;

    // ---- coalesced streaming load: 8 x global_load_b128 (non-temporal) ------
    const float* base = x + ((size_t)(b * IMG + hc * 8) * (size_t)IMG)
                          + (size_t)(wc * 8 + h4);
    f4 v[8];
#pragma unroll
    for (int r = 0; r < 8; ++r)
        v[r] = __builtin_nontemporal_load((const f4*)(base + (size_t)r * IMG));

    // ---- pass 1: tile max ----------------------------------------------------
    float mx = -FLT_MAX;
#pragma unroll
    for (int r = 0; r < 8; ++r) {
        mx = fmaxf(fmaxf(mx, fmaxf(v[r].x, v[r].y)), fmaxf(v[r].z, v[r].w));
    }
    mx = fmaxf(mx, __shfl_xor(mx, 16, 32));
    const float mb = mx * LOG2E;

    // ---- pass 2: e_i = exp(x_i - mx), inclusive cumsum, f16 pack ------------
    v16h a0, a1, onesh;
#pragma unroll
    for (int i = 0; i < 16; ++i) onesh[i] = (_Float16)1.0f;

    float xa[32];       // this lane's 32 x-values in scan order
    float cs[32];       // inclusive cumsum of e (f32, monotone nondecreasing)
    float c = 0.0f;
#pragma unroll
    for (int r = 0; r < 8; ++r) {
        const float xs[4] = {v[r].x, v[r].y, v[r].z, v[r].w};
#pragma unroll
        for (int cc = 0; cc < 4; ++cc) {
            const int k = r * 4 + cc;                       // local slot 0..31
            float xv = xs[cc];
            xa[k] = xv;
            float e = __builtin_amdgcn_exp2f(
                          __builtin_fmaf(xv, LOG2E, -mb));  // (0,1]
            if (k < 16) a0[k] = (_Float16)e; else a1[k - 16] = (_Float16)e;
            c += e;
            cs[k] = c;
        }
    }
    const float locTot = c;
    const float parTot = __shfl_xor(locTot, 16, 32);
    const float Stot   = locTot + parTot;                   // exact f32 total

    // ---- softmax denominator on the matrix pipe (for the LSE) ---------------
    // Row M = lane%16 = tile; B = ones so K-slot order is irrelevant. Two
    // chained 16x16x32 WMMAs accumulate all 64 exp-values per tile in f32.
    v8f acc = {};
    acc = __builtin_amdgcn_wmma_f32_16x16x32_f16(
        false, a0, false, onesh, (short)0, acc, false, false);
    acc = __builtin_amdgcn_wmma_f32_16x16x32_f16(
        false, a1, false, onesh, (short)0, acc, false, false);

    // ---- pass 3: inverse-CDF categorical draw (one uniform per tile) --------
    // choice = #(global cumsum < T), T = u*Stot. Scan order: half0's 32 slots
    // then half1's. Per-lane threshold T' = T - offset keeps compares local.
    float ucat = mix_u01(((unsigned)t << 1) + 0x68E31DA4u);     // same both halves
    float Tth  = __builtin_fmaf(ucat, Stot, half ? -parTot : 0.0f);

    int   cnt = 0;
    float sel = xa[0];                                      // x at local crossing
#pragma unroll
    for (int k = 0; k < 32; ++k) {
        bool lt = cs[k] < Tth;                              // monotone in k
        cnt += lt ? 1 : 0;
        if (k < 31) sel = lt ? xa[k + 1] : sel;
    }
    int j = cnt + __shfl_xor(cnt, 16, 32);                  // global index 0..64
    j = (j > 63) ? 63 : j;                                  // u->1 rounding guard
    float oSel = __shfl_xor(sel, 16, 32);
    const int hsel = j >> 5;                                // which half owns it
    float selected = (hsel == half) ? sel : oSel;
    const int s  = j & 31;
    const int di = s >> 2;                                  // row in tile
    const int dj = hsel * 4 + (s & 3);                      // col in tile

    // ---- log-softmax at chosen index (WMMA row sum) -------------------------
    int idx = lane & 7;                                     // D: VGPR r = M=r/M=r+8
    float x0 = (idx & 1) ? acc[1] : acc[0];
    float x1 = (idx & 1) ? acc[3] : acc[2];
    float x2 = (idx & 1) ? acc[5] : acc[4];
    float x3 = (idx & 1) ? acc[7] : acc[6];
    float y0 = (idx & 2) ? x1 : x0;
    float y1 = (idx & 2) ? x3 : x2;
    float own     = (idx & 4) ? y1 : y0;
    float partner = __shfl_xor(own, 16, 32);
    float ssum    = (lane & 8) ? partner : own;

    float lse    = __builtin_fmaf(__builtin_amdgcn_logf(ssum), LN2, mx);
    float cat_lp = selected - lse;

    // ---- bernoulli acceptance + log-prob ------------------------------------
    float u2  = mix_u01((unsigned)t ^ 0x7F4A7C15u);
    float ens = __builtin_amdgcn_exp2f(-selected * LOG2E);  // exp(-sel)
    bool accpt = __builtin_fmaf(u2, ens, u2) < 1.0f;        // u < sigmoid(sel)
    // log_sigmoid(sel); log_sigmoid(-sel) = log_sigmoid(sel) - sel
    float lsg = -LN2 * __builtin_amdgcn_logf(1.0f + ens);
    float lp  = cat_lp + lsg - (accpt ? 0.0f : selected);

    // ---- outputs (one lane per tile, non-temporal streaming stores) ---------
    if (lane < 16) {
        f2 kp;
        kp.x = (float)(wc * 8 + dj);                        // cols
        kp.y = (float)(hc * 8 + di);                        // rows
        __builtin_nontemporal_store(kp, (f2*)out + t);              // kp_xy
        __builtin_nontemporal_store(lp, out + 2 * NTILES + t);      // log_probs
        __builtin_nontemporal_store(accpt ? 1.0f : 0.0f,
                                    out + 3 * NTILES + t);          // mask
    }
}

extern "C" void kernel_launch(void* const* d_in, const int* in_sizes, int n_in,
                              void* d_out, int out_size, void* d_ws, size_t ws_size,
                              hipStream_t stream) {
    (void)in_sizes; (void)n_in; (void)out_size; (void)d_ws; (void)ws_size;
    const float* x = (const float*)d_in[0];
    float* out = (float*)d_out;
    // 524288 tiles / 128 tiles-per-block = 4096 blocks, no tail.
    keypoint_sampler_kernel<<<dim3(NTILES / 128), dim3(256), 0, stream>>>(x, out);
}
